// BaseRNN_44787918962746
// MI455X (gfx1250) — matrix operand order repaired and checked by reading, not compiled, not run on hardware
//
#include <hip/hip_runtime.h>
#include <hip/hip_bf16.h>

// Problem constants (from reference): B=64, T=512, V=32000, E=512, H=512, L=2
#define BB 64
#define TT 512
#define EE 512
#define HH 512

typedef __bf16 v16bf __attribute__((ext_vector_type(16)));
typedef float  v8f   __attribute__((ext_vector_type(8)));
typedef unsigned int v4u  __attribute__((ext_vector_type(4)));
typedef int          v8i_ __attribute__((ext_vector_type(8)));
typedef int          v4i_ __attribute__((ext_vector_type(4)));

__device__ __forceinline__ unsigned short f2bf(float f) {
    union { float f; unsigned int u; } cv; cv.f = f;
    unsigned int u = cv.u;
    unsigned int r = u + 0x7FFFu + ((u >> 16) & 1u);   // round-to-nearest-even
    return (unsigned short)(r >> 16);
}

// low 32 bits of the LDS-aperture flat address = byte offset in LDS
__device__ __forceinline__ unsigned lds_off(const void* p) {
    return (unsigned)(unsigned long long)p;
}

// ---------------------------------------------------------------------------
// TDM: 1-D linear load (contiguous block) global -> LDS, 8-byte elements.
// D# per CDNA5 ISA. count=1, type=2("image").
// ---------------------------------------------------------------------------
__device__ __forceinline__ void tdm_load_linear8(const void* gsrc, unsigned ldsByte,
                                                 unsigned nElem8B) {
    unsigned long long ga = (unsigned long long)gsrc;
    v4u g0;
    g0.x = 1u;                                   // count=1, user mode
    g0.y = ldsByte;                              // lds_addr
    g0.z = (unsigned)(ga & 0xffffffffu);         // global_addr[31:0]   (bits 95:64)
    g0.w = (unsigned)(ga >> 32) | (2u << 30);    // global_addr[56:32] | type=2
    v8i_ g1;
    g1[0] = (3 << 16);                           // workgroup_mask=0, data_size=3 (8B)
    g1[1] = (int)(nElem8B << 16);                // tensor_dim0[15:0]  (bits 63:48)
    g1[2] = (int)((nElem8B >> 16) | (1u << 16)); // tensor_dim0 hi | tensor_dim1=1
    g1[3] = (int)(nElem8B << 16);                // tensor_dim1 hi(0) | tile_dim0
    g1[4] = 1;                                   // tile_dim1=1, tile_dim2=0
    g1[5] = (int)nElem8B;                        // tensor_dim0_stride lo
    g1[6] = 0;                                   // stride0 hi | stride1 lo
    g1[7] = 0;
    v4i_ z4 = {0, 0, 0, 0};
    v8i_ z8 = {0, 0, 0, 0, 0, 0, 0, 0};
    __builtin_amdgcn_tensor_load_to_lds(g0, g1, z4, z4, z8, 0);
}

// ---------------------------------------------------------------------------
// TDM gather: 16 rows of `rowLen` f32 elements selected by 16-bit row indices
// (descriptor groups 2/3), global -> LDS.  tile_dim1 = #valid indices = 16.
// tokens must be uniform (SGPR) values.
// ---------------------------------------------------------------------------
__device__ __forceinline__ void tdm_gather16_f32(const float* base, unsigned ldsByte,
                                                 unsigned rowLen, unsigned nRows,
                                                 int tokLane /* lane<16: row index */) {
    unsigned long long ga = (unsigned long long)base;
    v4u g0;
    g0.x = 1u | (1u << 31);                      // count=1, gather_mode=1, 16-bit idx
    g0.y = ldsByte;
    g0.z = (unsigned)(ga & 0xffffffffu);
    g0.w = (unsigned)(ga >> 32) | (2u << 30);
    v8i_ g1;
    g1[0] = (2 << 16);                           // data_size=2 (4B)
    g1[1] = (int)(rowLen << 16);                 // tensor_dim0 lo
    g1[2] = (int)((rowLen >> 16) | (nRows << 16)); // tensor_dim0 hi | tensor_dim1 lo
    g1[3] = (int)(rowLen << 16);                 // tensor_dim1 hi(0) | tile_dim0
    g1[4] = 16;                                  // tile_dim1 = #valid indices
    g1[5] = (int)rowLen;                         // tensor_dim0_stride lo
    g1[6] = 0;
    g1[7] = 0;
    v4i_ g2, g3;
    #pragma unroll
    for (int i = 0; i < 4; ++i) {
        int lo = __builtin_amdgcn_readlane(tokLane, 2 * i);
        int hi = __builtin_amdgcn_readlane(tokLane, 2 * i + 1);
        g2[i] = (lo & 0xffff) | (hi << 16);
    }
    #pragma unroll
    for (int i = 0; i < 4; ++i) {
        int lo = __builtin_amdgcn_readlane(tokLane, 8 + 2 * i);
        int hi = __builtin_amdgcn_readlane(tokLane, 8 + 2 * i + 1);
        g3[i] = (lo & 0xffff) | (hi << 16);
    }
    v8i_ z8 = {0, 0, 0, 0, 0, 0, 0, 0};
    __builtin_amdgcn_tensor_load_to_lds(g0, g1, g2, g3, z8, 0);
}

// ---------------------------------------------------------------------------
// Pack the four 512x512 f32 weight matrices into bf16 WMMA B-operand tiles.
// Per matrix layout: [nt(32)][kt(16)][lane(32)][j(16)]  (1 KB per 16x32 tile)
//   n = nt*16 + (lane&15), k = kt*32 + (lane>>4)*8 + (j&7) + (j>>3)*16, value = W[n*512 + k]
// Matrix order: 0=W_ih[0], 1=W_hh[0], 2=W_ih[1], 3=W_hh[1]
// ---------------------------------------------------------------------------
__global__ __launch_bounds__(256) void pack_weights(
    const float* __restrict__ W_ih, const float* __restrict__ W_hh,
    unsigned short* __restrict__ wpack)
{
    const float* srcs[4] = { W_ih, W_hh, W_ih + HH * EE, W_hh + HH * HH };
    for (int idx = blockIdx.x * blockDim.x + threadIdx.x; idx < 4 * 262144;
         idx += gridDim.x * blockDim.x) {
        int q    = idx >> 18;
        int p    = idx & 262143;
        int j    = p & 15;
        int lane = (p >> 4) & 31;
        int kt   = (p >> 9) & 15;
        int nt   = p >> 13;
        int n = nt * 16 + (lane & 15);
        int k = kt * 32 + (lane >> 4) * 8 + (j & 7) + (j >> 3) * 16;
        wpack[idx] = f2bf(srcs[q][n * 512 + k]);
    }
}

__global__ __launch_bounds__(256) void zero_ws(unsigned int* __restrict__ p, int nwords) {
    for (int i = blockIdx.x * blockDim.x + threadIdx.x; i < nwords;
         i += gridDim.x * blockDim.x) p[i] = 0u;
}

// ---------------------------------------------------------------------------
// Barrier across the 8 column-slice workgroups of one batch-group.
// CDNA5 cluster barrier as the HW fast path; agent-scope atomic as the
// unconditional correctness path (monotonic epochs, reset by zero_ws).
// ---------------------------------------------------------------------------
__device__ __forceinline__ void group_barrier(int* cnt, int epoch) {
    __builtin_amdgcn_fence(__ATOMIC_RELEASE, "agent");
    __syncthreads();
    if (threadIdx.x == 0)
        __hip_atomic_fetch_add(cnt, 1, __ATOMIC_RELEASE, __HIP_MEMORY_SCOPE_AGENT);
    __builtin_amdgcn_s_cluster_barrier();   // s_barrier_signal -3 / s_barrier_wait -3
    if (threadIdx.x == 0) {
        while (__hip_atomic_load(cnt, __ATOMIC_ACQUIRE, __HIP_MEMORY_SCOPE_AGENT) < 8 * epoch)
            __builtin_amdgcn_s_sleep(4);
    }
    __syncthreads();
    __builtin_amdgcn_fence(__ATOMIC_ACQUIRE, "agent");
}

// ---------------------------------------------------------------------------
// Persistent recurrent kernel.  Grid: 32 WGs x 128 threads (4 waves).
//   g = blockIdx>>3 : batch group, rows b in [16g,16g+16)  (one WMMA M-tile)
//   w = blockIdx&7  : hidden column slice [64w,64w+64); wave owns one N-tile.
// LDS: 256 KB weight slices (TDM-loaded) + 16 KB packed-A emb + 32 KB f32
// gather stage (TDM gather prefetches step t+1 during step t's WMMAs).
// ---------------------------------------------------------------------------
__global__ __launch_bounds__(128) void rnn_persistent(
    const int* __restrict__ x, const int* __restrict__ lengths,
    const float* __restrict__ emb,
    const float* __restrict__ b_ih, const float* __restrict__ b_hh,
    const unsigned short* __restrict__ wpack,
    unsigned short* __restrict__ h0buf, unsigned short* __restrict__ h1buf,
    int* __restrict__ arrive, float* __restrict__ out)
{
    extern __shared__ __align__(128) unsigned char smem[];
    unsigned short* lw     = (unsigned short*)smem;      // [4][4 nt][16 kt][32][16]
    unsigned short* lemb   = lw + 4 * 32768;             // [16 kt][32][16]  (A layout)
    float*          estage = (float*)(lemb + 8192);      // [16 rows][512]   (gather dst)

    const int tid  = threadIdx.x;
    const int g    = blockIdx.x >> 3;
    const int w    = blockIdx.x & 7;
    const int wave = tid >> 5;
    const int lane = tid & 31;
    const int hi8  = (lane >> 4) * 8;
    const int n_lane = w * 64 + wave * 16 + (lane & 15);

    // token for this lane's batch row at t=0 (lanes 0..15 of each wave)
    int tok0 = (lane < 16) ? x[(g * 16 + lane) * TT + 0] : 0;

    // ---- one-time: TDM-stream this WG's weight slices + t=0 emb gather ----
    if (wave == 0) {
        #pragma unroll
        for (int q = 0; q < 4; ++q)
            tdm_load_linear8(wpack + q * 262144 + (w * 4) * 8192,
                             lds_off(lw + q * 32768), 8192);
        tdm_gather16_f32(emb, lds_off(estage), EE, 32000, tok0);
    }

    // per-thread persistent hidden state (f32, matches WMMA C/D layout)
    float hreg0[8], hreg1[8];
    int   len8[8];
    #pragma unroll
    for (int i = 0; i < 8; ++i) {
        hreg0[i] = 0.f; hreg1[i] = 0.f;
        len8[i]  = lengths[g * 16 + i + hi8];
    }
    const float bias0 = b_ih[n_lane]      + b_hh[n_lane];
    const float bias1 = b_ih[HH + n_lane] + b_hh[HH + n_lane];

    // packed-A write coordinates for this thread's h values (k = n_lane)
    const int kt2   = n_lane >> 5;
    const int koff  = n_lane & 31;
    const int pl_hi = (koff >> 3) & 1;
    const int j2    = (koff & 7) + ((koff >> 4) & 1) * 8;

    for (int t = 0; t < TT; ++t) {
        const int cur = t & 1, nxt = cur ^ 1;

        // ---- wait for TDM gather of this step's emb rows, then repack ----
        if (wave == 0) __builtin_amdgcn_s_wait_tensorcnt(0);
        __syncthreads();

        // estage (16 x 512 f32, row-major) -> lemb (bf16 WMMA-A layout)
        for (int c = tid; c < 1024; c += 128) {
            int kt   = c >> 6;
            int lp   = (c >> 1) & 31;
            int hseg = c & 1;
            int m    = lp & 15;
            int hi   = lp >> 4;
            int k0   = kt * 32 + hi * 8 + hseg * 16;
            const float4* e4 = (const float4*)(estage + m * EE + k0);
            float4 fa = e4[0], fb = e4[1];
            unsigned int w0 = f2bf(fa.x) | ((unsigned int)f2bf(fa.y) << 16);
            unsigned int w1 = f2bf(fa.z) | ((unsigned int)f2bf(fa.w) << 16);
            unsigned int w2 = f2bf(fb.x) | ((unsigned int)f2bf(fb.y) << 16);
            unsigned int w3 = f2bf(fb.z) | ((unsigned int)f2bf(fb.w) << 16);
            ((uint4*)lemb)[c] = make_uint4(w0, w1, w2, w3);
        }
        __syncthreads();

        // ---- prefetch: issue TDM gather for step t+1 (overlaps the WMMAs) ----
        if (wave == 0 && t + 1 < TT) {
            int tokn = (lane < 16) ? x[(g * 16 + lane) * TT + (t + 1)] : 0;
            tdm_gather16_f32(emb, lds_off(estage), EE, 32000, tokn);
        }

        // ================= layer 0 =================
        v8f acc;
        #pragma unroll
        for (int i = 0; i < 8; ++i) acc[i] = bias0;

        const unsigned short* h0c = h0buf + cur * 32768 + g * 8192;
        #pragma unroll
        for (int kt = 0; kt < 16; ++kt) {   // inp @ W_ih0^T
            v16bf a  = *(const v16bf*)(lemb + (kt * 32 + lane) * 16);
            v16bf bm = *(const v16bf*)(lw + 0 * 32768 + ((wave * 16 + kt) * 32 + lane) * 16);
            acc = __builtin_amdgcn_wmma_f32_16x16x32_bf16(false, a, false, bm,
                                                          (short)0, acc, false, false);
        }
        #pragma unroll
        for (int kt = 0; kt < 16; ++kt) {   // h0_prev @ W_hh0^T
            v16bf a  = *(const v16bf*)(h0c + (kt * 32 + lane) * 16);
            v16bf bm = *(const v16bf*)(lw + 1 * 32768 + ((wave * 16 + kt) * 32 + lane) * 16);
            acc = __builtin_amdgcn_wmma_f32_16x16x32_bf16(false, a, false, bm,
                                                          (short)0, acc, false, false);
        }
        unsigned short* h0n = h0buf + nxt * 32768 + g * 8192;
        #pragma unroll
        for (int i = 0; i < 8; ++i) {
            float hv = tanhf(acc[i]);
            float o  = (t < len8[i]) ? hv : hreg0[i];
            hreg0[i] = o;
            int m = i + hi8;
            h0n[(kt2 * 32 + m + pl_hi * 16) * 16 + j2] = f2bf(o);
        }
        group_barrier(&arrive[g], 2 * t + 1);

        // ================= layer 1 =================
        v8f acc1;
        #pragma unroll
        for (int i = 0; i < 8; ++i) acc1[i] = bias1;

        const unsigned short* h1c = h1buf + cur * 32768 + g * 8192;
        #pragma unroll
        for (int kt = 0; kt < 16; ++kt) {   // h0_new @ W_ih1^T
            v16bf a  = *(const v16bf*)(h0n + (kt * 32 + lane) * 16);
            v16bf bm = *(const v16bf*)(lw + 2 * 32768 + ((wave * 16 + kt) * 32 + lane) * 16);
            acc1 = __builtin_amdgcn_wmma_f32_16x16x32_bf16(false, a, false, bm,
                                                           (short)0, acc1, false, false);
        }
        #pragma unroll
        for (int kt = 0; kt < 16; ++kt) {   // h1_prev @ W_hh1^T
            v16bf a  = *(const v16bf*)(h1c + (kt * 32 + lane) * 16);
            v16bf bm = *(const v16bf*)(lw + 3 * 32768 + ((wave * 16 + kt) * 32 + lane) * 16);
            acc1 = __builtin_amdgcn_wmma_f32_16x16x32_bf16(false, a, false, bm,
                                                           (short)0, acc1, false, false);
        }
        unsigned short* h1n = h1buf + nxt * 32768 + g * 8192;
        #pragma unroll
        for (int i = 0; i < 8; ++i) {
            float hv = tanhf(acc1[i]);
            float o  = (t < len8[i]) ? hv : hreg1[i];
            hreg1[i] = o;
            int m = i + hi8;
            h1n[(kt2 * 32 + m + pl_hi * 16) * 16 + j2] = f2bf(o);
        }
        group_barrier(&arrive[g], 2 * t + 2);
    }

    // ---- outputs: (hidden[:, -1], hidden)  -> [B*H] ++ [B*L*H] f32 ----
    float* out1 = out;            // (B, H)    = final layer-1 hidden
    float* out2 = out + BB * HH;  // (B, L, H)
    #pragma unroll
    for (int i = 0; i < 8; ++i) {
        int b = g * 16 + i + hi8;
        out1[b * HH + n_lane]           = hreg1[i];
        out2[(b * 2 + 0) * HH + n_lane] = hreg0[i];
        out2[(b * 2 + 1) * HH + n_lane] = hreg1[i];
    }
}

// ---------------------------------------------------------------------------
extern "C" void kernel_launch(void* const* d_in, const int* in_sizes, int n_in,
                              void* d_out, int out_size, void* d_ws, size_t ws_size,
                              hipStream_t stream) {
    (void)in_sizes; (void)n_in; (void)out_size; (void)ws_size;
    const int*   x       = (const int*)d_in[0];
    const int*   lengths = (const int*)d_in[1];
    const float* emb     = (const float*)d_in[2];
    const float* W_ih    = (const float*)d_in[3];
    const float* W_hh    = (const float*)d_in[4];
    const float* b_ih    = (const float*)d_in[5];
    const float* b_hh    = (const float*)d_in[6];

    // workspace layout (~2.27 MB)
    unsigned short* wpack = (unsigned short*)d_ws;   // 4 * 262144 bf16 (2 MB)
    unsigned short* h0buf = wpack + 4 * 262144;      // 2 phases * 32768 bf16
    unsigned short* h1buf = h0buf + 2 * 32768;       // 2 phases * 32768 bf16
    int*            arrive = (int*)(h1buf + 2 * 32768);   // 4 counters

    pack_weights<<<512, 256, 0, stream>>>(W_ih, W_hh, wpack);

    // zero hidden double-buffers + barrier counters every launch (graph-safe)
    const int zwords = (2 * 2 * 32768 * 2 + 16) / 4;  // h0buf+h1buf bytes + 16
    zero_ws<<<64, 256, 0, stream>>>((unsigned int*)h0buf, zwords);

    // LDS: 256 KB weights + 16 KB packed emb + 32 KB gather stage = 304 KB
    const size_t smem = (size_t)(4 * 32768 + 8192) * sizeof(unsigned short)
                      + (size_t)8192 * sizeof(float);
    rnn_persistent<<<32, 128, smem, stream>>>(x, lengths, emb, b_ih, b_hh,
                                              wpack, h0buf, h1buf, arrive,
                                              (float*)d_out);
}
